// BlockwiseTEDStudent_28707561407171
// MI455X (gfx1250) — compile-verified, matrix operand order checked by reading, not run on hardware
//
#include <hip/hip_runtime.h>
#include <hip/hip_bf16.h>

typedef __attribute__((ext_vector_type(16))) _Float16 v16h;
typedef __attribute__((ext_vector_type(8)))  _Float16 v8h;
typedef __attribute__((ext_vector_type(8)))  float    v8f;

#define B_TOT 4096
#define LSEQ  32
#define DM    256
#define DI    512
#define NL    4
// LDS strides (in elements); multiples of 8 halves to keep 16B alignment of fragments
#define LDF   264    // feat rows (f16), K used = 256
#define LDXZ  1032   // xz rows (f16), cols 0..511 = xc, 512..1023 = z
#define LDXD  52     // xdbl rows (f32), cols 0..47

struct P {
    // tokenizer
    const float *emb_dir, *emb_flags, *emb_proto, *fus_b, *fus_w;
    const float *iat_b, *iat_w, *len_b, *len_w, *ln_b, *ln_g;
    // layers (f32 small params)
    const float *A_log[NL], *Dp[NL], *conv_b[NL], *conv_w[NL], *dt_b[NL], *dt_w[NL];
    // layers (f16 GEMM weights, in d_ws)
    const _Float16 *in_h[NL], *x_h[NL], *out_h[NL];
    // final norm
    const float *norm_b, *norm_g;
    // heads, logical order e=0->exit8, 1->exit16, 2->exit32
    const float *cls_b1[3], *cls_b2[3], *cls_w1[3], *cls_w2[3];
    const float *cnf_b1[3], *cnf_b2[3], *cnf_w1[3], *cnf_w2[3];
    const float *x;
    float *out;
};

__global__ void cvt_f16(const float* __restrict__ s, _Float16* __restrict__ d, int n) {
    int i = blockIdx.x * blockDim.x + threadIdx.x;
    if (i < n) d[i] = (_Float16)s[i];
}

// Load a 16x32 f16 A-operand fragment (also used as B-operand of Wt since
// A-layout(W) == B-layout(W^T)).  rowk = &M[row][k0]; off = 8*(lane>=16).
__device__ __forceinline__ v16h fragA(const _Float16* rowk, int off) {
    v8h lo = *(const v8h*)(rowk + off);
    v8h hi = *(const v8h*)(rowk + off + 16);
    v16h r;
#pragma unroll
    for (int i = 0; i < 8; ++i) { r[i] = lo[i]; r[8 + i] = hi[i]; }
    return r;
}

__device__ __forceinline__ float siluf(float v)     { return v / (1.f + __expf(-v)); }
__device__ __forceinline__ float softplusf(float v) { return (v > 20.f) ? v : __logf(1.f + __expf(v)); }

__device__ __forceinline__ void layer_norm_lds(_Float16* sF, float* sC,
                                               const float* g, const float* b, int tid) {
    if (tid < 32) {
        float mu = 0.f;
        for (int c = 0; c < DM; ++c) mu += (float)sF[tid * LDF + c];
        mu *= (1.f / DM);
        float var = 0.f;
        for (int c = 0; c < DM; ++c) { float d = (float)sF[tid * LDF + c] - mu; var += d * d; }
        sC[1600 + tid] = mu;
        sC[1632 + tid] = rsqrtf(var * (1.f / DM) + 1e-5f);
    }
    __syncthreads();
    for (int i = tid; i < 32 * DM; i += 256) {
        int t = i >> 8, c = i & 255;
        float v = ((float)sF[t * LDF + c] - sC[1600 + t]) * sC[1632 + t] * g[c] + b[c];
        sF[t * LDF + c] = (_Float16)v;
    }
    __syncthreads();
}

__global__ __launch_bounds__(256) void mamba_fused(P p) {
    __shared__ __align__(16) _Float16 sF[32 * LDF];    // 16.5 KB  current features (f16)
    __shared__ __align__(16) _Float16 sXZ[32 * LDXZ];  // 64.5 KB  xz / xc / gated-y
    __shared__ __align__(16) float    sC[32 * LDXD];   //  6.5 KB  xdbl / reps / heads / LN stats
    __shared__ __align__(16) float    sX[LSEQ * 5];    //  640 B   raw input tile (async-staged)

    const int tid  = threadIdx.x;
    const int lane = tid & 31;
    const int wv   = tid >> 5;
    const int col  = lane & 15;
    const int half = lane >> 4;
    const int off  = half * 8;
    const int seq  = blockIdx.x;

    // ---- async global->LDS staging of the input tile (ASYNCcnt path) ----
    {
        const float* xg = p.x + (size_t)seq * LSEQ * 5;
        if (tid < 40) {                                  // 40 x 16B = 640B
            unsigned ldsb = (unsigned)(uintptr_t)(&sX[tid * 4]);
            unsigned voff = tid * 16;
            asm volatile("global_load_async_to_lds_b128 %0, %1, %2"
                         :: "v"(ldsb), "v"(voff),
                            "s"((unsigned long long)(uintptr_t)xg)
                         : "memory");
        }
        asm volatile("s_wait_asynccnt 0x0" ::: "memory");
        __syncthreads();
    }

    // ---------------- tokenizer ----------------
    {
        float* sFu = (float*)sXZ;                    // 32 x 136 fused features
        for (int i = tid; i < 32 * 136; i += 256) {
            int t = i / 136, j = i - t * 136;
            const float* xf = sX + t * 5;
            float v;
            if (j < 32)       { int id = (int)xf[0]; id = id < 0 ? 0 : (id > 255 ? 255 : id); v = p.emb_proto[id * 32 + j]; }
            else if (j < 64)  { int c = j - 32;  v = xf[1] * p.len_w[c] + p.len_b[c]; }
            else if (j < 96)  { int id = (int)xf[2]; id = id < 0 ? 0 : (id > 63 ? 63 : id);  v = p.emb_flags[id * 32 + (j - 64)]; }
            else if (j < 128) { int c = j - 96;  v = xf[3] * p.iat_w[c] + p.iat_b[c]; }
            else              { int id = (int)xf[4]; id = id < 0 ? 0 : (id > 1 ? 1 : id);    v = p.emb_dir[id * 8 + (j - 128)]; }
            sFu[t * 136 + j] = v;
        }
        __syncthreads();
        for (int i = tid; i < 32 * DM; i += 256) {
            int t = i >> 8, c = i & 255;
            float a = p.fus_b[c];
            const float* w = p.fus_w + c * 136;
            const float* f = sFu + t * 136;
            for (int j = 0; j < 136; ++j) a += f[j] * w[j];
            sF[t * LDF + c] = (_Float16)a;
        }
        __syncthreads();
        layer_norm_lds(sF, sC, p.ln_g, p.ln_b, tid);
    }

    // ---------------- mamba layers ----------------
    for (int l = 0; l < NL; ++l) {
        // --- GEMM1: xz(32x1024) = feat(32x256) @ in_proj_w^T ---
        // one wave per nt column; both M-tiles share each B fragment (2 wmma / B-load)
        {
            const _Float16* W = p.in_h[l];
            const _Float16* a0 = sF + col * LDF;
            const _Float16* a1 = sF + (16 + col) * LDF;
            for (int nt = wv; nt < 64; nt += 8) {
                const _Float16* brow = W + (nt * 16 + col) * DM;
                __builtin_prefetch(W + ((nt + 8) * 16 + col) * DM, 0, 1);
                v8f acc0 = {}, acc1 = {};
#pragma unroll
                for (int kt = 0; kt < 8; ++kt) {
                    v16h b  = fragA(brow + kt * 32, off);
                    v16h x0 = fragA(a0 + kt * 32, off);
                    v16h x1 = fragA(a1 + kt * 32, off);
                    acc0 = __builtin_amdgcn_wmma_f32_16x16x32_f16(false, x0, false, b,
                                                                  (short)0, acc0, false, false);
                    acc1 = __builtin_amdgcn_wmma_f32_16x16x32_f16(false, x1, false, b,
                                                                  (short)0, acc1, false, false);
                }
#pragma unroll
                for (int r = 0; r < 8; ++r) {
                    sXZ[(half * 8 + r) * LDXZ + nt * 16 + col]        = (_Float16)acc0[r];
                    sXZ[(16 + half * 8 + r) * LDXZ + nt * 16 + col]   = (_Float16)acc1[r];
                }
            }
        }
        __syncthreads();

        // --- depthwise causal conv4 + silu, in place on xc ---
        {
#pragma unroll
            for (int i = 0; i < 2; ++i) {
                int d = tid * 2 + i;
                float w0 = p.conv_w[l][d * 4 + 0], w1 = p.conv_w[l][d * 4 + 1];
                float w2 = p.conv_w[l][d * 4 + 2], w3 = p.conv_w[l][d * 4 + 3];
                float cb = p.conv_b[l][d];
                float xm3 = 0.f, xm2 = 0.f, xm1 = 0.f;
                for (int t = 0; t < 32; ++t) {
                    float xt = (float)sXZ[t * LDXZ + d];
                    float v  = cb + w0 * xm3 + w1 * xm2 + w2 * xm1 + w3 * xt;
                    sXZ[t * LDXZ + d] = (_Float16)siluf(v);
                    xm3 = xm2; xm2 = xm1; xm1 = xt;
                }
            }
        }
        __syncthreads();

        // --- GEMM2: xdbl(32x48) = xc(32x512) @ x_proj_w^T ---
        if (wv < 3) {
            const _Float16* W = p.x_h[l];
            int nt = wv;
            const _Float16* a0 = sXZ + col * LDXZ;
            const _Float16* a1 = sXZ + (16 + col) * LDXZ;
            const _Float16* brow = W + (nt * 16 + col) * DI;
            v8f acc0 = {}, acc1 = {};
#pragma unroll
            for (int kt = 0; kt < 16; ++kt) {
                v16h b  = fragA(brow + kt * 32, off);
                v16h x0 = fragA(a0 + kt * 32, off);
                v16h x1 = fragA(a1 + kt * 32, off);
                acc0 = __builtin_amdgcn_wmma_f32_16x16x32_f16(false, x0, false, b,
                                                              (short)0, acc0, false, false);
                acc1 = __builtin_amdgcn_wmma_f32_16x16x32_f16(false, x1, false, b,
                                                              (short)0, acc1, false, false);
            }
#pragma unroll
            for (int r = 0; r < 8; ++r) {
                sC[(half * 8 + r) * LDXD + nt * 16 + col]      = acc0[r];
                sC[(16 + half * 8 + r) * LDXD + nt * 16 + col] = acc1[r];
            }
        }
        __syncthreads();

        // --- selective scan + dt_proj + gating; thread owns channels 2*tid, 2*tid+1 ---
        {
            float Areg[2][16], Wdt[2][16], hst[2][16], Dv[2], dtb[2];
#pragma unroll
            for (int i = 0; i < 2; ++i) {
                int d = tid * 2 + i;
                Dv[i]  = p.Dp[l][d];
                dtb[i] = p.dt_b[l][d];
#pragma unroll
                for (int s = 0; s < 16; ++s) {
                    Areg[i][s] = -__expf(p.A_log[l][d * 16 + s]);
                    Wdt[i][s]  = p.dt_w[l][d * 16 + s];
                    hst[i][s]  = 0.f;
                }
            }
            for (int t = 0; t < 32; ++t) {
                float dtv[16], Bv[16], Cv[16];
#pragma unroll
                for (int s = 0; s < 16; ++s) {
                    dtv[s] = sC[t * LDXD + s];
                    Bv[s]  = sC[t * LDXD + 16 + s];
                    Cv[s]  = sC[t * LDXD + 32 + s];
                }
#pragma unroll
                for (int i = 0; i < 2; ++i) {
                    int d = tid * 2 + i;
                    float acc = dtb[i];
#pragma unroll
                    for (int s = 0; s < 16; ++s) acc += dtv[s] * Wdt[i][s];
                    float delta = softplusf(acc);
                    float xv = (float)sXZ[t * LDXZ + d];
                    float dx = delta * xv, y = 0.f;
#pragma unroll
                    for (int s = 0; s < 16; ++s) {
                        float h = __expf(delta * Areg[i][s]) * hst[i][s] + dx * Bv[s];
                        hst[i][s] = h;
                        y += h * Cv[s];
                    }
                    y += xv * Dv[i];
                    float zv = (float)sXZ[t * LDXZ + 512 + d];
                    sXZ[t * LDXZ + d] = (_Float16)(y * siluf(zv));   // gated out_proj input
                }
            }
        }
        __syncthreads();

        // --- GEMM4: feat += gated(32x512) @ out_proj_w^T (residual fused) ---
        {
            const _Float16* W = p.out_h[l];
            const _Float16* a0 = sXZ + col * LDXZ;
            const _Float16* a1 = sXZ + (16 + col) * LDXZ;
            for (int nt = wv; nt < 16; nt += 8) {
                const _Float16* brow = W + (nt * 16 + col) * DI;
                __builtin_prefetch(W + ((nt + 8) * 16 + col) * DI, 0, 1);
                v8f acc0 = {}, acc1 = {};
#pragma unroll
                for (int kt = 0; kt < 16; ++kt) {
                    v16h b  = fragA(brow + kt * 32, off);
                    v16h x0 = fragA(a0 + kt * 32, off);
                    v16h x1 = fragA(a1 + kt * 32, off);
                    acc0 = __builtin_amdgcn_wmma_f32_16x16x32_f16(false, x0, false, b,
                                                                  (short)0, acc0, false, false);
                    acc1 = __builtin_amdgcn_wmma_f32_16x16x32_f16(false, x1, false, b,
                                                                  (short)0, acc1, false, false);
                }
#pragma unroll
                for (int r = 0; r < 8; ++r) {
                    int cc = nt * 16 + col;
                    int r0 = half * 8 + r, r1 = 16 + half * 8 + r;
                    float v0 = acc0[r] + (float)sF[r0 * LDF + cc];
                    float v1 = acc1[r] + (float)sF[r1 * LDF + cc];
                    sF[r0 * LDF + cc] = (_Float16)v0;
                    sF[r1 * LDF + cc] = (_Float16)v1;
                }
            }
        }
        __syncthreads();
        layer_norm_lds(sF, sC, p.norm_g, p.norm_b, tid);
    }

    // ---------------- prefix-mean reps + early-exit heads ----------------
    {
        int c = tid;                       // 256 threads == 256 channels
        float acc = 0.f;
        for (int t = 0; t < 32; ++t) {
            acc += (float)sF[t * LDF + c];
            if (t == 7)  sC[c]        = acc * (1.f / 8.f);
            if (t == 15) sC[256 + c]  = acc * (1.f / 16.f);
        }
        sC[512 + c] = acc * (1.f / 32.f);
    }
    __syncthreads();
    if (tid < 128) {
        int typ = tid >> 6, u = tid & 63;
        for (int e = 0; e < 3; ++e) {
            const float* w1 = typ ? p.cnf_w1[e] : p.cls_w1[e];
            const float* b1 = typ ? p.cnf_b1[e] : p.cls_b1[e];
            const float* rep = sC + e * 256;
            float a = b1[u];
            for (int c = 0; c < DM; ++c) a += rep[c] * w1[u * DM + c];
            sC[768 + e * 128 + typ * 64 + u] = a > 0.f ? a : 0.f;
        }
    }
    __syncthreads();
    if (tid < 3) {
        int e = tid;
        const float* hc = sC + 768 + e * 128;
        const float* hf = hc + 64;
        float l0 = p.cls_b2[e][0], l1 = p.cls_b2[e][1], cf = p.cnf_b2[e][0];
        for (int u = 0; u < 64; ++u) {
            l0 += hc[u] * p.cls_w2[e][u];
            l1 += hc[u] * p.cls_w2[e][64 + u];
            cf += hf[u] * p.cnf_w2[e][u];
        }
        cf = 1.f / (1.f + __expf(-cf));
        sC[1540 + e * 3 + 0] = l0;
        sC[1540 + e * 3 + 1] = l1;
        sC[1540 + e * 3 + 2] = cf;
    }
    __syncthreads();
    if (tid == 0) {
        float c8 = sC[1540 + 2], c16 = sC[1543 + 2];
        int sel = (c8 >= 0.9f) ? 0 : ((c16 >= 0.9f) ? 1 : 2);
        p.out[seq * 2 + 0]     = sC[1540 + sel * 3 + 0];
        p.out[seq * 2 + 1]     = sC[1540 + sel * 3 + 1];
        p.out[B_TOT * 2 + seq] = (float)(8 << sel);
    }
}

extern "C" void kernel_launch(void* const* d_in, const int* in_sizes, int n_in,
                              void* d_out, int out_size, void* d_ws, size_t ws_size,
                              hipStream_t stream) {
    (void)in_sizes; (void)n_in; (void)out_size; (void)ws_size;
    auto f = [&](int i) { return (const float*)d_in[i]; };

    P p;
    // heads flattened alphabetically: '16' @0, '32' @8, '8' @16; each dict:
    // cls_b1, cls_b2, cls_w1, cls_w2, conf_b1, conf_b2, conf_w1, conf_w2
    const int hb[3] = {16, 0, 8};   // logical exits 8,16,32
    for (int e = 0; e < 3; ++e) {
        int b = hb[e];
        p.cls_b1[e] = f(b + 0); p.cls_b2[e] = f(b + 1);
        p.cls_w1[e] = f(b + 2); p.cls_w2[e] = f(b + 3);
        p.cnf_b1[e] = f(b + 4); p.cnf_b2[e] = f(b + 5);
        p.cnf_w1[e] = f(b + 6); p.cnf_w2[e] = f(b + 7);
    }

    // layers @24: A_log, D, conv_b, conv_w, dt_proj_b, dt_proj_w, in_proj_w, out_proj_w, x_proj_w
    _Float16* wsH = (_Float16*)d_ws;
    const size_t LSTRIDE = 417792, XOFF = 262144, OOFF = 286720;
    for (int l = 0; l < NL; ++l) {
        int b = 24 + l * 9;
        p.A_log[l]  = f(b + 0); p.Dp[l]    = f(b + 1);
        p.conv_b[l] = f(b + 2); p.conv_w[l] = f(b + 3);
        p.dt_b[l]   = f(b + 4); p.dt_w[l]  = f(b + 5);
        const float* inw  = f(b + 6);
        const float* outw = f(b + 7);
        const float* xw   = f(b + 8);
        _Float16* ih = wsH + l * LSTRIDE;
        _Float16* xh = ih + XOFF;
        _Float16* oh = ih + OOFF;
        cvt_f16<<<(262144 + 255) / 256, 256, 0, stream>>>(inw,  ih, 262144);
        cvt_f16<<<(24576  + 255) / 256, 256, 0, stream>>>(xw,   xh, 24576);
        cvt_f16<<<(131072 + 255) / 256, 256, 0, stream>>>(outw, oh, 131072);
        p.in_h[l] = ih; p.x_h[l] = xh; p.out_h[l] = oh;
    }

    p.norm_b = f(60); p.norm_g = f(61);
    p.emb_dir = f(62); p.emb_flags = f(63); p.emb_proto = f(64);
    p.fus_b = f(65); p.fus_w = f(66);
    p.iat_b = f(67); p.iat_w = f(68);
    p.len_b = f(69); p.len_w = f(70);
    p.ln_b = f(71); p.ln_g = f(72);
    p.x = f(73);
    p.out = (float*)d_out;

    mamba_fused<<<B_TOT, 256, 0, stream>>>(p);
}
